// SelectiveMemoryLayer3_57543971832182
// MI455X (gfx1250) — matrix operand order repaired and checked by reading, not compiled, not run on hardware
//
#include <hip/hip_runtime.h>
#include <cmath>

// ---------------------------------------------------------------------------
// MI455X (gfx1250) implementation. wave32, WMMA f16 16x16x32 for all GEMMs,
// Tensor Data Mover (tensor_load_to_lds + s_wait_tensorcnt) for GEMM tile
// staging. Workspace requirement: ~1.4 GB.
// ---------------------------------------------------------------------------

typedef __attribute__((ext_vector_type(16))) _Float16 v16h;
typedef __attribute__((ext_vector_type(8)))  float    v8f;
typedef __attribute__((ext_vector_type(4)))  unsigned tdm4u;
typedef __attribute__((ext_vector_type(8)))  int      tdm8i;
typedef __attribute__((ext_vector_type(4)))  int      tdm4i;

union Frag16 { v16h v; uint4 u[2]; };
union U4H    { uint4 u; _Float16 h[8]; };

#if __has_builtin(__builtin_amdgcn_tensor_load_to_lds)
#define HAVE_TDM 1
#else
#define HAVE_TDM 0
#endif

// Problem constants (match reference)
constexpr int  B_ = 8, T_ = 128, L_ = 64, D_ = 768, I_ = 3072, H_ = 12, S_ = 8;
constexpr int  HD   = D_ / H_;           // 64
constexpr int  TOPK = 5;
constexpr int  NSEQ = B_ * T_;           // 1024
constexpr int  MTOK = NSEQ * L_;         // 65536 token rows
constexpr int  MZ   = NSEQ * (TOPK + 1); // 6144 mini rows
constexpr long TOK_OUT = (long)MTOK * D_;     // 50331648
constexpr long MEM_OUT = (long)B_ * S_ * D_;  // 49152

// Input index map (setup_inputs dict order, params flattened in insertion order)
enum {
  IN_TOKENS = 0, IN_MASK, IN_EDU, IN_SPK, IN_MEM,
  P_ATTN_WQ, P_ATTN_BQ, P_ATTN_WK, P_ATTN_BK, P_ATTN_WV, P_ATTN_BV, P_ATTN_WO, P_ATTN_BO,
  P_MA_WQ, P_MA_BQ, P_MA_WK, P_MA_BK, P_MA_WV, P_MA_BV, P_MA_WO, P_MA_BO,
  P_FFN_W1, P_FFN_B1, P_FFN_W2, P_FFN_B2,
  P_MF_W1, P_MF_B1, P_MF_W2, P_MF_B2,
  P_N1_G, P_N1_B, P_N2_G, P_N2_B, P_MN1_G, P_MN1_B, P_MN2_G, P_MN2_B,
  P_SPK_W1, P_SPK_B1, P_SPK_W2, P_SPK_B2,
  P_CNT_W1, P_CNT_B1, P_CNT_W2, P_CNT_B2,
  P_QPROJ, P_KPROJ,
  P_GRU_WIH, P_GRU_BIH, P_GRU_WHH, P_GRU_BHH
};

// ---------------------------------------------------------------------------
// TDM: DMA one 64-row x 32-half f16 tile (row stride = K elements) from global
// into LDS at lds_byte_off. HW padding of 4 DWORDs after every 16 DWORDs makes
// the LDS row stride 64B+16B = 80B = 40 halves (matches LDA below).
// D# bit layout per CDNA5 ISA ch.8 (group0 128b, group1 256b; groups 2/3 zero
// for a 2D tensor).
// ---------------------------------------------------------------------------
#if HAVE_TDM
__device__ __forceinline__ void tdm_load_tile_64x32_f16(
    const _Float16* gsrc, unsigned lds_byte_off, unsigned stride_elems) {
  unsigned long long ga = (unsigned long long)(uintptr_t)gsrc;
  tdm4u g0;
  g0[0] = 1u;                                            // count=1, user D#
  g0[1] = lds_byte_off;                                  // lds_addr  [63:32]
  g0[2] = (unsigned)ga;                                  // gaddr lo  [95:64]
  g0[3] = (unsigned)((ga >> 32) & 0x01FFFFFFull)         // gaddr hi  [120:96]
          | (2u << 30);                                  // type=2    [127:126]
  tdm8i g1;
  g1[0] = (int)0x06D10000u;  // wg_mask=0 | data_size=2B | pad_en | int=16dw | amt=4dw
  g1[1] = (int)(32u << 16);  // atomic_barrier_addr=0 | tensor_dim0 lo16 = 32
  g1[2] = (int)(64u << 16);  // tensor_dim0 hi16 = 0  | tensor_dim1 lo16 = 64
  g1[3] = (int)(32u << 16);  // tensor_dim1 hi16 = 0  | tile_dim0 = 32
  g1[4] = (int)64u;          // tile_dim1 = 64        | tile_dim2 = 0
  g1[5] = (int)stride_elems; // tensor_dim0_stride lo32 (elements)
  g1[6] = 0;                 // stride hi16 | tensor_dim1_stride lo16 (unused)
  g1[7] = 0;
  tdm4i gz = {0, 0, 0, 0};
#if defined(__clang_major__) && (__clang_major__ >= 23)
  tdm8i gz8 = {0, 0, 0, 0, 0, 0, 0, 0};
  __builtin_amdgcn_tensor_load_to_lds(g0, g1, gz, gz, gz8, 0);
#else
  __builtin_amdgcn_tensor_load_to_lds(g0, g1, gz, gz, 0);
#endif
}
#endif

// ---------------------------------------------------------------------------
// fp32 -> f16 convert
// ---------------------------------------------------------------------------
__global__ __launch_bounds__(256) void k_f2h(const float* __restrict__ s,
                                             _Float16* __restrict__ d, long n) {
  long i = (long)blockIdx.x * 256 + threadIdx.x;
  if (i < n) d[i] = (_Float16)s[i];
}

// ---------------------------------------------------------------------------
// Generic WMMA GEMM:  C[b] = act(A[b] (MxK, f16) @ W[b]^T (NxK, f16) + bias)
// 64x64 output tile per 128-thread block (4 waves), BK = 32.
// Tiles staged global->LDS by the Tensor Data Mover (wave 0 issues, waits
// TENSORcnt, workgroup barrier publishes). ACT: 0=none 1=gelu 2=tanh.
// ---------------------------------------------------------------------------
template <int ACT, bool W32, bool W16, bool HASB>
__global__ __launch_bounds__(128) void k_gemm(
    const _Float16* __restrict__ A, const _Float16* __restrict__ W,
    const float* __restrict__ bias, float* __restrict__ C32,
    _Float16* __restrict__ C16, int M, int N, int K,
    long sA, long sW, long sC) {
  A += (long)blockIdx.z * sA;
  W += (long)blockIdx.z * sW;
  const long cbase = (long)blockIdx.z * sC;

  const int tid = threadIdx.x, wave = tid >> 5, lane = tid & 31, l16 = lane & 15;
  const bool hi = lane >= 16;
  const int m0 = blockIdx.y * 64, n0 = blockIdx.x * 64;

  constexpr int LDA = 40;  // halves; 80B rows (64B data + 16B TDM pad)
  __shared__ __align__(16) _Float16 sAt[64 * LDA];
  __shared__ __align__(16) _Float16 sWt[64 * LDA];

  v8f acc[4] = {};

#if HAVE_TDM
  const unsigned ldsA = (unsigned)(uintptr_t)(void*)&sAt[0];
  const unsigned ldsW = (unsigned)(uintptr_t)(void*)&sWt[0];
#else
  const int lrow = tid >> 1;          // 0..63
  const int lcol = (tid & 1) * 16;    // 0 or 16 (halves)
#endif

  for (int k0 = 0; k0 < K; k0 += 32) {
#if HAVE_TDM
    __syncthreads();                  // all waves done reading previous tile
    if (wave == 0) {
      tdm_load_tile_64x32_f16(A + (long)m0 * K + k0, ldsA, (unsigned)K);
      tdm_load_tile_64x32_f16(W + (long)n0 * K + k0, ldsW, (unsigned)K);
      __builtin_amdgcn_s_wait_tensorcnt(0);
    }
    __syncthreads();                  // tiles visible to all waves
#else
    uint4 a0{0,0,0,0}, a1{0,0,0,0}, b0{0,0,0,0}, b1{0,0,0,0};
    if (m0 + lrow < M) {
      const _Float16* p = A + (long)(m0 + lrow) * K + k0 + lcol;
      a0 = *(const uint4*)p; a1 = *(const uint4*)(p + 8);
    }
    if (n0 + lrow < N) {
      const _Float16* p = W + (long)(n0 + lrow) * K + k0 + lcol;
      b0 = *(const uint4*)p; b1 = *(const uint4*)(p + 8);
    }
    __syncthreads();
    *(uint4*)&sAt[lrow * LDA + lcol]     = a0;
    *(uint4*)&sAt[lrow * LDA + lcol + 8] = a1;
    *(uint4*)&sWt[lrow * LDA + lcol]     = b0;
    *(uint4*)&sWt[lrow * LDA + lcol + 8] = b1;
    __syncthreads();
#endif

    // A fragment: row m = wave*16 + l16; lanes 0-15 hold K {0..7,16..23},
    // lanes 16-31 hold K {8..15,24..31} (per ISA 16-bit A layout)
    Frag16 af;
    const _Float16* ap = &sAt[(wave * 16 + l16) * LDA + (hi ? 8 : 0)];
    af.u[0] = *(const uint4*)ap;
    af.u[1] = *(const uint4*)(ap + 16);

#pragma unroll
    for (int n = 0; n < 4; ++n) {
      // B fragment: col n*16+l16; lanes 0-15 hold K {0..7,16..23}, lanes 16-31 K {8..15,24..31}
      Frag16 bf;
      const _Float16* bp = &sWt[(n * 16 + l16) * LDA + (hi ? 8 : 0)];
      bf.u[0] = *(const uint4*)bp;
      bf.u[1] = *(const uint4*)(bp + 16);
      acc[n] = __builtin_amdgcn_wmma_f32_16x16x32_f16(
          false, af.v, false, bf.v, (short)0, acc[n], false, false);
    }
  }

#pragma unroll
  for (int n = 0; n < 4; ++n) {
    int col = n0 + n * 16 + l16;
    if (col >= N) continue;
    float bv = HASB ? bias[col] : 0.0f;
#pragma unroll
    for (int r = 0; r < 8; ++r) {
      int row = m0 + wave * 16 + (hi ? 8 : 0) + r;
      if (row >= M) continue;
      float v = acc[n][r] + bv;
      if (ACT == 1) v = 0.5f * v * (1.0f + erff(v * 0.70710678118f));
      if (ACT == 2) v = tanhf(v);
      long off = cbase + (long)row * N + col;
      if (W32) C32[off] = v;
      if (W16) C16[off] = (_Float16)v;
    }
  }
}

// ---------------------------------------------------------------------------
// Token attention: one block per (head, sequence). L=64, hd=64.
// S = QK^T (WMMA), masked softmax in LDS, O = A·V (WMMA). f16 in/out.
// ---------------------------------------------------------------------------
__global__ __launch_bounds__(128) void k_attn_tok(
    const _Float16* __restrict__ Q, const _Float16* __restrict__ K,
    const _Float16* __restrict__ V, const float* __restrict__ mask,
    _Float16* __restrict__ O) {
  const int h = blockIdx.x, seq = blockIdx.y;
  const int tid = threadIdx.x, wave = tid >> 5, lane = tid & 31, l16 = lane & 15;
  const bool hi = lane >= 16;
  constexpr int LD = 72;
  __shared__ __align__(16) _Float16 sQ[64 * LD], sK[64 * LD], sVt[64 * LD];
  __shared__ float sS[64 * 65];

  const long rb = ((long)seq * 64) * D_ + (long)h * HD;
  {
    int r = tid >> 1, c0 = (tid & 1) * 32;
    const _Float16* qp = Q + rb + (long)r * D_ + c0;
    const _Float16* kp = K + rb + (long)r * D_ + c0;
    const _Float16* vp = V + rb + (long)r * D_ + c0;
#pragma unroll
    for (int j = 0; j < 4; ++j) {
      *(uint4*)&sQ[r * LD + c0 + j * 8] = *(const uint4*)(qp + j * 8);
      *(uint4*)&sK[r * LD + c0 + j * 8] = *(const uint4*)(kp + j * 8);
      U4H t; t.u = *(const uint4*)(vp + j * 8);
#pragma unroll
      for (int e = 0; e < 8; ++e) sVt[(c0 + j * 8 + e) * LD + r] = t.h[e];
    }
  }
  __syncthreads();

  // S = Q K^T, scale + key mask
  {
    v8f acc[4] = {};
#pragma unroll
    for (int k0 = 0; k0 < 64; k0 += 32) {
      Frag16 af;
      const _Float16* ap = &sQ[(wave * 16 + l16) * LD + k0 + (hi ? 8 : 0)];
      af.u[0] = *(const uint4*)ap;
      af.u[1] = *(const uint4*)(ap + 16);
#pragma unroll
      for (int n = 0; n < 4; ++n) {
        Frag16 bf;
        const _Float16* bp = &sK[(n * 16 + l16) * LD + k0 + (hi ? 8 : 0)];
        bf.u[0] = *(const uint4*)bp;
        bf.u[1] = *(const uint4*)(bp + 16);
        acc[n] = __builtin_amdgcn_wmma_f32_16x16x32_f16(
            false, af.v, false, bf.v, (short)0, acc[n], false, false);
      }
    }
#pragma unroll
    for (int n = 0; n < 4; ++n) {
      int col = n * 16 + l16;
      bool kvalid = mask[(long)seq * L_ + col] > 0.5f;
#pragma unroll
      for (int r = 0; r < 8; ++r) {
        int row = wave * 16 + (hi ? 8 : 0) + r;
        sS[row * 65 + col] = kvalid ? acc[n][r] * 0.125f : -1.0e9f;
      }
    }
  }
  __syncthreads();

  // row softmax -> f16 A written over sQ
  if (tid < 64) {
    float mx = -1e30f;
    for (int c = 0; c < 64; ++c) mx = fmaxf(mx, sS[tid * 65 + c]);
    float sm = 0.f;
    for (int c = 0; c < 64; ++c) sm += __expf(sS[tid * 65 + c] - mx);
    float inv = 1.0f / sm;
    for (int c = 0; c < 64; ++c)
      sQ[tid * LD + c] = (_Float16)(__expf(sS[tid * 65 + c] - mx) * inv);
  }
  __syncthreads();

  // O = A · V  (B operand = V transposed: sVt[d][n])
  {
    v8f acc[4] = {};
#pragma unroll
    for (int k0 = 0; k0 < 64; k0 += 32) {
      Frag16 af;
      const _Float16* ap = &sQ[(wave * 16 + l16) * LD + k0 + (hi ? 8 : 0)];
      af.u[0] = *(const uint4*)ap;
      af.u[1] = *(const uint4*)(ap + 16);
#pragma unroll
      for (int n = 0; n < 4; ++n) {
        Frag16 bf;
        const _Float16* bp = &sVt[(n * 16 + l16) * LD + k0 + (hi ? 8 : 0)];
        bf.u[0] = *(const uint4*)bp;
        bf.u[1] = *(const uint4*)(bp + 16);
        acc[n] = __builtin_amdgcn_wmma_f32_16x16x32_f16(
            false, af.v, false, bf.v, (short)0, acc[n], false, false);
      }
    }
#pragma unroll
    for (int n = 0; n < 4; ++n) {
      int col = n * 16 + l16;
#pragma unroll
      for (int r = 0; r < 8; ++r) {
        int row = wave * 16 + (hi ? 8 : 0) + r;
        O[rb + (long)row * D_ + col] = (_Float16)acc[n][r];
      }
    }
  }
}

// ---------------------------------------------------------------------------
// x = LN(res + y) * g + b ; write fp32 (+optional f16). One block per row, D=768.
// ---------------------------------------------------------------------------
__global__ __launch_bounds__(256) void k_add_ln(
    const float* __restrict__ res, const float* __restrict__ y,
    const float* __restrict__ g, const float* __restrict__ bb,
    float* __restrict__ o32, _Float16* __restrict__ o16) {
  const long row = blockIdx.x;
  const int tid = threadIdx.x;
  __shared__ float red[256];
  const long base = row * D_;
  float v0 = res[base + tid]       + y[base + tid];
  float v1 = res[base + tid + 256] + y[base + tid + 256];
  float v2 = res[base + tid + 512] + y[base + tid + 512];
  red[tid] = v0 + v1 + v2;
  __syncthreads();
  for (int st = 128; st; st >>= 1) { if (tid < st) red[tid] += red[tid + st]; __syncthreads(); }
  float mu = red[0] * (1.0f / 768.0f);
  __syncthreads();
  float d0 = v0 - mu, d1 = v1 - mu, d2 = v2 - mu;
  red[tid] = d0 * d0 + d1 * d1 + d2 * d2;
  __syncthreads();
  for (int st = 128; st; st >>= 1) { if (tid < st) red[tid] += red[tid + st]; __syncthreads(); }
  float inv = rsqrtf(red[0] * (1.0f / 768.0f) + 1e-5f);
  float r0 = d0 * inv * g[tid]       + bb[tid];
  float r1 = d1 * inv * g[tid + 256] + bb[tid + 256];
  float r2 = d2 * inv * g[tid + 512] + bb[tid + 512];
  o32[base + tid]       = r0;
  o32[base + tid + 256] = r1;
  o32[base + tid + 512] = r2;
  if (o16) {
    o16[base + tid]       = (_Float16)r0;
    o16[base + tid + 256] = (_Float16)r1;
    o16[base + tid + 512] = (_Float16)r2;
  }
}

// ---------------------------------------------------------------------------
// Masked mean pooling over L=64 tokens -> new_edu (fp32 + f16)
// ---------------------------------------------------------------------------
__global__ __launch_bounds__(256) void k_pool(
    const float* __restrict__ tok, const float* __restrict__ mask,
    float* __restrict__ ne32, _Float16* __restrict__ ne16) {
  const long i = blockIdx.x;  // (b*T + t)
  float msum = 0.f;
  for (int l = 0; l < L_; ++l) msum += mask[i * L_ + l];
  const float inv = 1.0f / fmaxf(msum, 1e-6f);
  for (int d = threadIdx.x; d < D_; d += 256) {
    float s = 0.f;
    for (int l = 0; l < L_; ++l)
      s += tok[(i * L_ + l) * D_ + d] * mask[i * L_ + l];
    float v = s * inv;
    ne32[i * D_ + d] = v;
    ne16[i * D_ + d] = (_Float16)v;
  }
}

// spk_log = tanh_hidden @ spk_W2^T + b2  (dot per row)
__global__ __launch_bounds__(256) void k_spk2(
    const _Float16* __restrict__ hR, const float* __restrict__ w2,
    const float* __restrict__ b2, float* __restrict__ out) {
  const long i = blockIdx.x;
  const int tid = threadIdx.x;
  __shared__ float red[256];
  float s = 0.f;
  for (int d = tid; d < D_; d += 256) s += (float)hR[i * D_ + d] * w2[d];
  red[tid] = s;
  __syncthreads();
  for (int st = 128; st; st >>= 1) { if (tid < st) red[tid] += red[tid + st]; __syncthreads(); }
  if (tid == 0) out[i] = red[0] + b2[0];
}

// ---------------------------------------------------------------------------
// keep-mask, n_keep, has_past, top-5 selection, kmask. One thread per (b,t).
// ---------------------------------------------------------------------------
__global__ __launch_bounds__(256) void k_topk(
    const float* __restrict__ scores, const float* __restrict__ spl,
    const int* __restrict__ spk, int* __restrict__ idx,
    float* __restrict__ nkf, int* __restrict__ hp, int* __restrict__ km) {
  int i = blockIdx.x * 256 + threadIdx.x;
  if (i >= NSEQ) return;
  int b = i / T_, t = i % T_;
  int myspk = spk[i];
  unsigned long long kb0 = 0, kb1 = 0;
  int nk = 0; bool anyv = false;
  for (int s = 0; s < T_; ++s) {
    bool valid = (s < t) && (spk[b * T_ + s] == myspk);
    anyv |= valid;
    float sg = 1.0f / (1.0f + __expf(-spl[b * T_ + s]));
    if (valid && sg > 0.5f) {
      nk++;
      if (s < 64) kb0 |= 1ull << s; else kb1 |= 1ull << (s - 64);
    }
  }
  const float* sc = scores + ((long)b * T_ + t) * T_;
  unsigned long long c0 = 0, c1 = 0;
  for (int j = 0; j < TOPK; ++j) {
    float best = -3.0e38f; int bi = 0;
    for (int s = 0; s < T_; ++s) {
      bool keep = (s < 64) ? ((kb0 >> s) & 1) : ((kb1 >> (s - 64)) & 1);
      bool ch = (s < 64) ? ((c0 >> s) & 1) : ((c1 >> (s - 64)) & 1);
      if (ch) continue;
      float v = keep ? sc[s] : -1.0e9f;
      if (v > best) { best = v; bi = s; }
    }
    idx[i * TOPK + j] = bi;
    if (bi < 64) c0 |= 1ull << bi; else c1 |= 1ull << (bi - 64);
  }
  nkf[i] = (float)nk;
  hp[i] = anyv ? 1 : 0;
  km[i * 6] = 1;
  for (int j = 0; j < TOPK; ++j) km[i * 6 + 1 + j] = (j < nk) ? 1 : 0;
}

// Build chosen = concat(new_edu[t], gathered top-k rows)  (fp32 + f16)
__global__ __launch_bounds__(256) void k_gather(
    const float* __restrict__ ne32, const int* __restrict__ idx,
    float* __restrict__ ch32, _Float16* __restrict__ ch16) {
  const int i = blockIdx.x, slot = blockIdx.y;
  const int b = i / T_;
  long src = (slot == 0) ? (long)i : (long)b * T_ + idx[i * TOPK + slot - 1];
  long dst = (long)i * 6 + slot;
  for (int d = threadIdx.x; d < D_; d += 256) {
    float v = ne32[src * D_ + d];
    ch32[dst * D_ + d] = v;
    ch16[dst * D_ + d] = (_Float16)v;
  }
}

// ---------------------------------------------------------------------------
// Mini attention over 6 slots: scalar (negligible FLOPs). One block per seq.
// ---------------------------------------------------------------------------
__global__ __launch_bounds__(128) void k_attn_mini(
    const _Float16* __restrict__ Q, const _Float16* __restrict__ K,
    const _Float16* __restrict__ V, const int* __restrict__ km,
    _Float16* __restrict__ O) {
  const int i = blockIdx.x, tid = threadIdx.x;
  if (tid >= H_ * 6) return;
  const int h = tid / 6, qp = tid % 6;
  const _Float16* qr = Q + ((long)i * 6 + qp) * D_ + h * HD;
  float a[6]; float mx = -1e30f;
  for (int s = 0; s < 6; ++s) {
    const _Float16* kr = K + ((long)i * 6 + s) * D_ + h * HD;
    float d = 0.f;
    for (int e = 0; e < HD; ++e) d += (float)qr[e] * (float)kr[e];
    d *= 0.125f;
    if (!km[i * 6 + s]) d = -1.0e9f;
    a[s] = d; mx = fmaxf(mx, d);
  }
  float sm = 0.f;
  for (int s = 0; s < 6; ++s) { a[s] = __expf(a[s] - mx); sm += a[s]; }
  const float inv = 1.0f / sm;
  for (int e = 0; e < HD; ++e) {
    float o = 0.f;
    for (int s = 0; s < 6; ++s)
      o += a[s] * (float)V[((long)i * 6 + s) * D_ + h * HD + e];
    O[((long)i * 6 + qp) * D_ + h * HD + e] = (_Float16)(o * inv);
  }
}

// summ = z[:, 0]
__global__ __launch_bounds__(256) void k_summ(const _Float16* __restrict__ z16,
                                              _Float16* __restrict__ s16) {
  const long i = blockIdx.x;
  for (int d = threadIdx.x; d < D_; d += 256)
    s16[i * D_ + d] = z16[(i * 6) * D_ + d];
}

// ---------------------------------------------------------------------------
// GRU scan: one block per independent (b, spk) chain. gi precomputed by WMMA.
// ---------------------------------------------------------------------------
__global__ __launch_bounds__(256) void k_gru(
    const float* __restrict__ mem_in, const _Float16* __restrict__ Whh,
    const float* __restrict__ bhh, const float* __restrict__ gi,
    const int* __restrict__ spk, const int* __restrict__ hp,
    const float* __restrict__ nkf, float* __restrict__ mem_out) {
  const int b = blockIdx.x / S_, s = blockIdx.x % S_;
  const int tid = threadIdx.x;
  __shared__ float h[D_];
  __shared__ float gh[3 * D_];
  for (int d = tid; d < D_; d += 256) h[d] = mem_in[((long)b * S_ + s) * D_ + d];
  __syncthreads();
  for (int t = 0; t < T_; ++t) {
    if (spk[b * T_ + t] != s) continue;            // block-uniform
    const long i = (long)b * T_ + t;
    if (!(hp[i] && nkf[i] > 0.5f)) continue;       // block-uniform
    for (int n = tid; n < 3 * D_; n += 256) {
      const _Float16* wr = Whh + (long)n * D_;
      float acc = bhh[n];
      for (int k = 0; k < D_; ++k) acc += (float)wr[k] * h[k];
      gh[n] = acc;
    }
    __syncthreads();
    for (int d = tid; d < D_; d += 256) {
      float ir = gi[i * (3 * D_) + d];
      float iz = gi[i * (3 * D_) + D_ + d];
      float in = gi[i * (3 * D_) + 2 * D_ + d];
      float r = 1.0f / (1.0f + __expf(-(ir + gh[d])));
      float z = 1.0f / (1.0f + __expf(-(iz + gh[D_ + d])));
      float n2 = tanhf(in + r * gh[2 * D_ + d]);
      h[d] = (1.0f - z) * n2 + z * h[d];
    }
    __syncthreads();
  }
  for (int d = tid; d < D_; d += 256) mem_out[((long)b * S_ + s) * D_ + d] = h[d];
}

// pen = n_keep at last flat index with has_past (scan semantics), else 0.
__global__ __launch_bounds__(256) void k_pen(const int* __restrict__ hp,
                                             const float* __restrict__ nkf,
                                             float* __restrict__ out) {
  __shared__ int red[256];
  int best = -1;
  for (int i = threadIdx.x; i < NSEQ; i += 256) if (hp[i]) best = i;
  red[threadIdx.x] = best;
  __syncthreads();
  for (int st = 128; st; st >>= 1) {
    if (threadIdx.x < st) red[threadIdx.x] = max(red[threadIdx.x], red[threadIdx.x + st]);
    __syncthreads();
  }
  if (threadIdx.x == 0) out[0] = (red[0] >= 0) ? nkf[red[0]] : 0.0f;
}

// ---------------------------------------------------------------------------
// Host orchestration
// ---------------------------------------------------------------------------
extern "C" void kernel_launch(void* const* d_in, const int* in_sizes, int n_in,
                              void* d_out, int out_size, void* d_ws, size_t ws_size,
                              hipStream_t stream) {
  (void)in_sizes; (void)n_in; (void)out_size; (void)ws_size;

  const float* tokens = (const float*)d_in[IN_TOKENS];
  const float* mask   = (const float*)d_in[IN_MASK];
  const int*   spk    = (const int*)d_in[IN_SPK];
  const float* mem0   = (const float*)d_in[IN_MEM];
  auto F = [&](int i) { return (const float*)d_in[i]; };

  float* out = (float*)d_out;
  float* out_tok = out;
  float* out_mem = out + TOK_OUT;
  float* out_pen = out + TOK_OUT + MEM_OUT;

  char* ws = (char*)d_ws;
  size_t off = 0;
  auto alloc = [&](size_t bytes) {
    size_t o = off; off += (bytes + 255) & ~(size_t)255; return o;
  };
  auto H16P = [&](size_t o) { return (_Float16*)(ws + o); };
  auto F32P = [&](size_t o) { return (float*)(ws + o); };
  auto I32P = [&](size_t o) { return (int*)(ws + o); };

  // Big buffers
  const size_t oX16 = alloc((size_t)MTOK * D_ * 2);  // activation f16
  const size_t oQ16 = alloc((size_t)MTOK * D_ * 2);
  const size_t oK16 = alloc((size_t)MTOK * D_ * 2);
  const size_t oV16 = alloc((size_t)MTOK * D_ * 2);
  const size_t oA16 = alloc((size_t)MTOK * D_ * 2);  // attention/O f16
  const size_t oY32 = alloc((size_t)MTOK * D_ * 4);  // gemm fp32 out
  const size_t oX32 = alloc((size_t)MTOK * D_ * 4);  // post-LN1 fp32
  const size_t oH16 = alloc((size_t)MTOK * I_ * 2);  // ffn hidden f16
  // f16 weights
  const size_t oWaQ = alloc((size_t)D_ * D_ * 2), oWaK = alloc((size_t)D_ * D_ * 2);
  const size_t oWaV = alloc((size_t)D_ * D_ * 2), oWaO = alloc((size_t)D_ * D_ * 2);
  const size_t oWf1 = alloc((size_t)I_ * D_ * 2), oWf2 = alloc((size_t)D_ * I_ * 2);
  const size_t oWmQ = alloc((size_t)D_ * D_ * 2), oWmK = alloc((size_t)D_ * D_ * 2);
  const size_t oWmV = alloc((size_t)D_ * D_ * 2), oWmO = alloc((size_t)D_ * D_ * 2);
  const size_t oWg1 = alloc((size_t)I_ * D_ * 2), oWg2 = alloc((size_t)D_ * I_ * 2);
  const size_t oWsp = alloc((size_t)D_ * D_ * 2);
  const size_t oWqp = alloc((size_t)D_ * D_ * 2), oWkp = alloc((size_t)D_ * D_ * 2);
  const size_t oWih = alloc((size_t)3 * D_ * D_ * 2), oWhh = alloc((size_t)3 * D_ * D_ * 2);
  // Small buffers
  const size_t oNE32 = alloc((size_t)NSEQ * D_ * 4);
  const size_t oNE16 = alloc((size_t)NSEQ * D_ * 2);
  const size_t oQE16 = alloc((size_t)NSEQ * D_ * 2);
  const size_t oKE16 = alloc((size_t)NSEQ * D_ * 2);
  const size_t oSC   = alloc((size_t)B_ * T_ * T_ * 4);
  const size_t oSPH  = alloc((size_t)NSEQ * D_ * 2);
  const size_t oSPL  = alloc((size_t)NSEQ * 4);
  const size_t oIDX  = alloc((size_t)NSEQ * TOPK * 4);
  const size_t oNK   = alloc((size_t)NSEQ * 4);
  const size_t oHP   = alloc((size_t)NSEQ * 4);
  const size_t oKM   = alloc((size_t)NSEQ * 6 * 4);
  const size_t oCH32 = alloc((size_t)MZ * D_ * 4);
  const size_t oCH16 = alloc((size_t)MZ * D_ * 2);
  const size_t oGI   = alloc((size_t)NSEQ * 3 * D_ * 4);
  const size_t oSUM  = alloc((size_t)NSEQ * D_ * 2);
  // reuse of big buffers for the mini layer
  const size_t oMZ32 = oX32, oMZ16 = oX16;

  auto conv = [&](int in_idx, size_t cnt, size_t dst) {
    k_f2h<<<dim3((unsigned)((cnt + 255) / 256)), 256, 0, stream>>>(
        F(in_idx), H16P(dst), (long)cnt);
  };

  // ---- weight + activation conversion (fp32 -> f16) ----
  conv(IN_TOKENS, (size_t)MTOK * D_, oX16);
  conv(P_ATTN_WQ, (size_t)D_ * D_, oWaQ); conv(P_ATTN_WK, (size_t)D_ * D_, oWaK);
  conv(P_ATTN_WV, (size_t)D_ * D_, oWaV); conv(P_ATTN_WO, (size_t)D_ * D_, oWaO);
  conv(P_FFN_W1, (size_t)I_ * D_, oWf1);  conv(P_FFN_W2, (size_t)D_ * I_, oWf2);
  conv(P_MA_WQ, (size_t)D_ * D_, oWmQ);   conv(P_MA_WK, (size_t)D_ * D_, oWmK);
  conv(P_MA_WV, (size_t)D_ * D_, oWmV);   conv(P_MA_WO, (size_t)D_ * D_, oWmO);
  conv(P_MF_W1, (size_t)I_ * D_, oWg1);   conv(P_MF_W2, (size_t)D_ * I_, oWg2);
  conv(P_SPK_W1, (size_t)D_ * D_, oWsp);
  conv(P_QPROJ, (size_t)D_ * D_, oWqp);   conv(P_KPROJ, (size_t)D_ * D_, oWkp);
  conv(P_GRU_WIH, (size_t)3 * D_ * D_, oWih);
  conv(P_GRU_WHH, (size_t)3 * D_ * D_, oWhh);

  const dim3 gTokD(D_ / 64, MTOK / 64, 1);   // (12, 1024)
  const dim3 gTokI(I_ / 64, MTOK / 64, 1);   // (48, 1024)

  // ---- token transformer layer ----
  k_gemm<0, false, true, true><<<gTokD, 128, 0, stream>>>(
      H16P(oX16), H16P(oWaQ), F(P_ATTN_BQ), nullptr, H16P(oQ16), MTOK, D_, D_, 0, 0, 0);
  k_gemm<0, false, true, true><<<gTokD, 128, 0, stream>>>(
      H16P(oX16), H16P(oWaK), F(P_ATTN_BK), nullptr, H16P(oK16), MTOK, D_, D_, 0, 0, 0);
  k_gemm<0, false, true, true><<<gTokD, 128, 0, stream>>>(
      H16P(oX16), H16P(oWaV), F(P_ATTN_BV), nullptr, H16P(oV16), MTOK, D_, D_, 0, 0, 0);
  k_attn_tok<<<dim3(H_, NSEQ), 128, 0, stream>>>(
      H16P(oQ16), H16P(oK16), H16P(oV16), mask, H16P(oA16));
  k_gemm<0, true, false, true><<<gTokD, 128, 0, stream>>>(
      H16P(oA16), H16P(oWaO), F(P_ATTN_BO), F32P(oY32), nullptr, MTOK, D_, D_, 0, 0, 0);
  k_add_ln<<<MTOK, 256, 0, stream>>>(
      tokens, F32P(oY32), F(P_N1_G), F(P_N1_B), F32P(oX32), H16P(oX16));
  k_gemm<1, false, true, true><<<gTokI, 128, 0, stream>>>(
      H16P(oX16), H16P(oWf1), F(P_FFN_B1), nullptr, H16P(oH16), MTOK, I_, D_, 0, 0, 0);
  k_gemm<0, true, false, true><<<gTokD, 128, 0, stream>>>(
      H16P(oH16), H16P(oWf2), F(P_FFN_B2), F32P(oY32), nullptr, MTOK, D_, I_, 0, 0, 0);
  k_add_ln<<<MTOK, 256, 0, stream>>>(
      F32P(oX32), F32P(oY32), F(P_N2_G), F(P_N2_B), out_tok, H16P(oX16));

  // ---- pooling, scores, speaker gate ----
  k_pool<<<NSEQ, 256, 0, stream>>>(out_tok, mask, F32P(oNE32), H16P(oNE16));
  const dim3 gEdu(D_ / 64, NSEQ / 64, 1);    // (12, 16)
  k_gemm<0, false, true, false><<<gEdu, 128, 0, stream>>>(
      H16P(oNE16), H16P(oWqp), nullptr, nullptr, H16P(oQE16), NSEQ, D_, D_, 0, 0, 0);
  k_gemm<0, false, true, false><<<gEdu, 128, 0, stream>>>(
      H16P(oNE16), H16P(oWkp), nullptr, nullptr, H16P(oKE16), NSEQ, D_, D_, 0, 0, 0);
  k_gemm<0, true, false, false><<<dim3(T_ / 64, T_ / 64, B_), 128, 0, stream>>>(
      H16P(oQE16), H16P(oKE16), nullptr, F32P(oSC), nullptr, T_, T_, D_,
      (long)T_ * D_, (long)T_ * D_, (long)T_ * T_);
  k_gemm<2, false, true, true><<<gEdu, 128, 0, stream>>>(
      H16P(oNE16), H16P(oWsp), F(P_SPK_B1), nullptr, H16P(oSPH), NSEQ, D_, D_, 0, 0, 0);
  k_spk2<<<NSEQ, 256, 0, stream>>>(H16P(oSPH), F(P_SPK_W2), F(P_SPK_B2), F32P(oSPL));
  k_topk<<<NSEQ / 256, 256, 0, stream>>>(
      F32P(oSC), F32P(oSPL), spk, I32P(oIDX), F32P(oNK), I32P(oHP), I32P(oKM));
  k_gather<<<dim3(NSEQ, 6), 256, 0, stream>>>(
      F32P(oNE32), I32P(oIDX), F32P(oCH32), H16P(oCH16));

  // ---- mini transformer layer over (TOPK+1)=6 slots ----
  const dim3 gMzD(D_ / 64, MZ / 64, 1);      // (12, 96)
  const dim3 gMzI(I_ / 64, MZ / 64, 1);      // (48, 96)
  k_gemm<0, false, true, true><<<gMzD, 128, 0, stream>>>(
      H16P(oCH16), H16P(oWmQ), F(P_MA_BQ), nullptr, H16P(oQ16), MZ, D_, D_, 0, 0, 0);
  k_gemm<0, false, true, true><<<gMzD, 128, 0, stream>>>(
      H16P(oCH16), H16P(oWmK), F(P_MA_BK), nullptr, H16P(oK16), MZ, D_, D_, 0, 0, 0);
  k_gemm<0, false, true, true><<<gMzD, 128, 0, stream>>>(
      H16P(oCH16), H16P(oWmV), F(P_MA_BV), nullptr, H16P(oV16), MZ, D_, D_, 0, 0, 0);
  k_attn_mini<<<NSEQ, 128, 0, stream>>>(
      H16P(oQ16), H16P(oK16), H16P(oV16), I32P(oKM), H16P(oA16));
  k_gemm<0, true, false, true><<<gMzD, 128, 0, stream>>>(
      H16P(oA16), H16P(oWmO), F(P_MA_BO), F32P(oY32), nullptr, MZ, D_, D_, 0, 0, 0);
  k_add_ln<<<MZ, 256, 0, stream>>>(
      F32P(oCH32), F32P(oY32), F(P_MN1_G), F(P_MN1_B), F32P(oMZ32), H16P(oMZ16));
  k_gemm<1, false, true, true><<<gMzI, 128, 0, stream>>>(
      H16P(oMZ16), H16P(oWg1), F(P_MF_B1), nullptr, H16P(oH16), MZ, I_, D_, 0, 0, 0);
  k_gemm<0, true, false, true><<<gMzD, 128, 0, stream>>>(
      H16P(oH16), H16P(oWg2), F(P_MF_B2), F32P(oY32), nullptr, MZ, D_, I_, 0, 0, 0);
  k_add_ln<<<MZ, 256, 0, stream>>>(
      F32P(oMZ32), F32P(oY32), F(P_MN2_G), F(P_MN2_B), F32P(oCH32), H16P(oCH16));
  k_summ<<<NSEQ, 256, 0, stream>>>(H16P(oCH16), H16P(oSUM));

  // ---- GRU: precompute gi via WMMA, then 64 independent chain scans ----
  k_gemm<0, true, false, true><<<dim3(3 * D_ / 64, NSEQ / 64, 1), 128, 0, stream>>>(
      H16P(oSUM), H16P(oWih), F(P_GRU_BIH), F32P(oGI), nullptr, NSEQ, 3 * D_, D_, 0, 0, 0);
  k_gru<<<B_ * S_, 256, 0, stream>>>(
      mem0, H16P(oWhh), F(P_GRU_BHH), F32P(oGI), spk, I32P(oHP), F32P(oNK), out_mem);
  k_pen<<<1, 256, 0, stream>>>(I32P(oHP), F32P(oNK), out_pen);
}
